// ODEFunc_72842645340708
// MI455X (gfx1250) — matrix operand order repaired and checked by reading, not compiled, not run on hardware
//
#include <hip/hip_runtime.h>
#include <hip/hip_bf16.h>
#include <math.h>

// Problem constants (match reference)
#define BB 8
#define NN 5000
#define PP 256
#define QQ 256
#define EE 160000
#define ROWS (BB * NN)   // 40000
#define DIN 512          // P+Q

typedef __attribute__((ext_vector_type(16))) __bf16 bf16x16;
typedef __attribute__((ext_vector_type(8)))  float  f32x8;
typedef __attribute__((ext_vector_type(4)))  float  f32x4;

union FragAB { bf16x16 v; f32x4 f[2]; };

__device__ __forceinline__ float softplusf(float x) {
  return (x > 20.0f) ? x : log1pf(__expf(x));
}

// ---------------------------------------------------------------------------
// K0: convert 5 weight matrices (each 256x512 f32, row-major [out,in]) to one
// bf16 blob: [W_A | W_Fc | W_Fh | W_Gc | W_Z], each 131072 elements.
// ---------------------------------------------------------------------------
__global__ void __launch_bounds__(256)
k_convert_w(const float* __restrict__ wfc, const float* __restrict__ wfh,
            const float* __restrict__ wgc, const float* __restrict__ wz,
            const float* __restrict__ wa, __bf16* __restrict__ wcat) {
  int i = blockIdx.x * 256 + threadIdx.x;
  if (i >= 5 * 131072) return;
  int m = i >> 17;          // which matrix
  int r = i & 131071;
  const float* p = (m == 0) ? wa : (m == 1) ? wfc : (m == 2) ? wfh
                 : (m == 3) ? wgc : wz;
  wcat[i] = (__bf16)p[r];
}

// K0b: pack biases [b_Fc | b_Fh | b_Gc | b_Z] (1024 floats)
__global__ void __launch_bounds__(256)
k_convert_b(const float* __restrict__ bfc, const float* __restrict__ bfh,
            const float* __restrict__ bgc, const float* __restrict__ bz,
            float* __restrict__ bcat) {
  int i = blockIdx.x * 256 + threadIdx.x;
  if (i >= 1024) return;
  int m = i >> 8, r = i & 255;
  const float* p = (m == 0) ? bfc : (m == 1) ? bfh : (m == 2) ? bgc : bz;
  bcat[i] = p[r];
}

// ---------------------------------------------------------------------------
// K1: degree count (float, one add per edge)
// ---------------------------------------------------------------------------
__global__ void __launch_bounds__(256)
k_degree(const int* __restrict__ dst, float* __restrict__ deg) {
  int e = blockIdx.x * 256 + threadIdx.x;
  if (e >= EE) return;
  atomicAdd(&deg[dst[e]], 1.0f);
}

// ---------------------------------------------------------------------------
// K2: scatter-add h[b, src, q] into sum[dst, b, q]. One block per edge,
// thread t owns column q=t across all 8 batches.
// ---------------------------------------------------------------------------
__global__ void __launch_bounds__(256)
k_scatter(const float* __restrict__ u, const int* __restrict__ src,
          const int* __restrict__ dst, float* __restrict__ sum) {
  int e = blockIdx.x;
  int s = src[e], d = dst[e];
  int q = threadIdx.x;
  float* base = sum + (size_t)d * (BB * QQ);
#pragma unroll
  for (int b = 0; b < BB; ++b) {
    float v = u[((size_t)b * NN + s) * DIN + PP + q];
    atomicAdd(base + b * QQ + q, v);
  }
}

// ---------------------------------------------------------------------------
// K3: copy c part of u into bf16 activation buffer u_ = [c | h_]
// ---------------------------------------------------------------------------
__global__ void __launch_bounds__(256)
k_copy_c(const float* __restrict__ u, __bf16* __restrict__ ubf) {
  int i = blockIdx.x * 256 + threadIdx.x;   // ROWS*256 elements
  if (i >= ROWS * PP) return;
  int r = i >> 8, j = i & 255;
  ubf[(size_t)r * DIN + j] = (__bf16)u[(size_t)r * DIN + j];
}

// ---------------------------------------------------------------------------
// K4: stage-1 GEMM: h_ = softplus([h | mean] @ W_A^T + b_A), write bf16
// into ubf[:, 256:512]. 64 rows per workgroup, 8 waves (4 row x 2 col),
// each wave: 16 rows x 128 cols = 8 WMMA accumulators, K=512 in 16 steps.
// Dynamic LDS: 64*512 bf16 A-tile (64 KB).
// ---------------------------------------------------------------------------
__global__ void __launch_bounds__(256)
k_stage1(const float* __restrict__ u, const float* __restrict__ sum,
         const float* __restrict__ deg, const __bf16* __restrict__ WA,
         const float* __restrict__ bA, __bf16* __restrict__ ubf) {
  extern __shared__ __bf16 lds_a[];               // [64][512]
  const int tid = threadIdx.x;
  const int rowbase = blockIdx.x * 64;

  // Build A tile: cols 0..255 = h (from u), cols 256..511 = neighbor mean
  for (int i = tid; i < 64 * DIN; i += 256) {
    int r = i >> 9, col = i & 511;
    int g = rowbase + r;
    float x;
    if (col < QQ) {
      x = u[(size_t)g * DIN + PP + col];
    } else {
      int b = g / NN, n = g - b * NN;
      float d = deg[n];
      d = (d < 1.0f) ? 1.0f : d;
      x = sum[(size_t)n * (BB * QQ) + b * QQ + (col - QQ)] * (1.0f / d);
    }
    lds_a[i] = (__bf16)x;
  }
  __syncthreads();

  const int w = tid >> 5, lane = tid & 31;
  const int sel = lane >> 4, l16 = lane & 15;
  const int mbase = (w >> 1) * 16;        // 0,16,32,48
  const int nbase = (w & 1) * 128;        // 0,128

  f32x8 acc[8];
#pragma unroll
  for (int nt = 0; nt < 8; ++nt)
    acc[nt] = f32x8{0.f, 0.f, 0.f, 0.f, 0.f, 0.f, 0.f, 0.f};

  for (int kb = 0; kb < DIN; kb += 32) {
    FragAB a;
    const __bf16* ap = &lds_a[(mbase + l16) * DIN + kb + sel * 8];
    a.f[0] = *reinterpret_cast<const f32x4*>(ap);
    a.f[1] = *reinterpret_cast<const f32x4*>(ap + 16);
#pragma unroll
    for (int nt = 0; nt < 8; ++nt) {
      int n = nbase + nt * 16 + l16;
      const __bf16* bp = &WA[(size_t)n * DIN + kb + sel * 8];
      FragAB bb;
      bb.f[0] = *reinterpret_cast<const f32x4*>(bp);
      bb.f[1] = *reinterpret_cast<const f32x4*>(bp + 16);
      acc[nt] = __builtin_amdgcn_wmma_f32_16x16x32_bf16(
          false, a.v, false, bb.v, (short)0, acc[nt], false, false);
    }
  }

  // Epilogue: bias + softplus -> bf16 store into h_ slot
#pragma unroll
  for (int nt = 0; nt < 8; ++nt) {
    int coln = nbase + nt * 16 + l16;
    float bias = bA[coln];
#pragma unroll
    for (int v = 0; v < 8; ++v) {
      int r = rowbase + mbase + v + sel * 8;
      float val = softplusf(acc[nt][v] + bias);
      ubf[(size_t)r * DIN + QQ + coln] = (__bf16)val;
    }
  }
}

// ---------------------------------------------------------------------------
// K5: stage-2 GEMM + fused epilogue.  One workgroup = 16 rows x 1024 cols
// (Fc|Fh|Gc|Zt).  8 waves, wave w owns cols [w*128, w*128+128).  Results go
// to LDS (16x1024 f32, 64 KB dynamic), then 256 threads do the elementwise
// combine, per-row projection (shfl_xor + LDS reduce), and write d_out.
// ---------------------------------------------------------------------------
__global__ void __launch_bounds__(256)
k_stage2(const float* __restrict__ u, const __bf16* __restrict__ ubf,
         const __bf16* __restrict__ W2, const float* __restrict__ b2,
         float* __restrict__ out) {
  extern __shared__ float lds_o[];                // [16][1024]
  __shared__ float s_red[17];
  const int tid = threadIdx.x;
  const int rowbase = blockIdx.x * 16;

  const int w = tid >> 5, lane = tid & 31;
  const int sel = lane >> 4, l16 = lane & 15;
  const int nbase = w * 128;

  f32x8 acc[8];
#pragma unroll
  for (int nt = 0; nt < 8; ++nt)
    acc[nt] = f32x8{0.f, 0.f, 0.f, 0.f, 0.f, 0.f, 0.f, 0.f};

  const __bf16* arow = ubf + (size_t)(rowbase + l16) * DIN;
  for (int kb = 0; kb < DIN; kb += 32) {
    FragAB a;
    const __bf16* ap = arow + kb + sel * 8;
    a.f[0] = *reinterpret_cast<const f32x4*>(ap);
    a.f[1] = *reinterpret_cast<const f32x4*>(ap + 16);
#pragma unroll
    for (int nt = 0; nt < 8; ++nt) {
      int n = nbase + nt * 16 + l16;                // 0..1023
      const __bf16* bp = &W2[(size_t)n * DIN + kb + sel * 8];
      FragAB bb;
      bb.f[0] = *reinterpret_cast<const f32x4*>(bp);
      bb.f[1] = *reinterpret_cast<const f32x4*>(bp + 16);
      acc[nt] = __builtin_amdgcn_wmma_f32_16x16x32_bf16(
          false, a.v, false, bb.v, (short)0, acc[nt], false, false);
    }
  }

  // bias + activation -> LDS
#pragma unroll
  for (int nt = 0; nt < 8; ++nt) {
    int col = nbase + nt * 16 + l16;
    float bias = b2[col];
    int m = col >> 8;                               // 0=Fc,1=Fh,2=Gc,3=Zt
#pragma unroll
    for (int v = 0; v < 8; ++v) {
      int r = v + sel * 8;
      float val = acc[nt][v] + bias;
      val = (m == 3) ? tanhf(val) : softplusf(val);
      lds_o[r * 1024 + col] = val;
    }
  }
  __syncthreads();

  // Fused elementwise + per-row projection
  const int j = tid;                                // column 0..255
  for (int r = 0; r < 16; ++r) {
    size_t g = (size_t)(rowbase + r);
    float c  = u[g * DIN + j];
    float h  = u[g * DIN + PP + j];
    float Fc = lds_o[r * 1024 + j];
    float Fh = lds_o[r * 1024 + 256 + j];
    float Gc = lds_o[r * 1024 + 512 + j];
    float Zt = lds_o[r * 1024 + 768 + j];
    float dc = -Fc * c + Gc * Zt;
    out[g * DIN + PP + j] = -Fh * h;               // dh

    float pd = dc * c, pc = c * c;
#pragma unroll
    for (int off = 16; off > 0; off >>= 1) {
      pd += __shfl_xor(pd, off, 32);
      pc += __shfl_xor(pc, off, 32);
    }
    if (lane == 0) { s_red[w] = pd; s_red[8 + w] = pc; }
    __syncthreads();
    if (tid == 0) {
      float sd = 0.f, sc = 0.f;
#pragma unroll
      for (int i = 0; i < 8; ++i) { sd += s_red[i]; sc += s_red[8 + i]; }
      s_red[16] = sd / sc;
    }
    __syncthreads();
    float proj = s_red[16];
    out[g * DIN + j] = dc - proj * c;              // corrected dc
    __syncthreads();
  }
}

// ---------------------------------------------------------------------------
// Host-side launch (graph-capture safe: only kernel launches + memsetAsync)
// ---------------------------------------------------------------------------
extern "C" void kernel_launch(void* const* d_in, const int* in_sizes, int n_in,
                              void* d_out, int out_size, void* d_ws, size_t ws_size,
                              hipStream_t stream) {
  const float* u    = (const float*)d_in[0];
  const int*   src  = (const int*)d_in[1];
  const int*   dst  = (const int*)d_in[2];
  // d_in[3] = t (unused)
  const float* W_Fc = (const float*)d_in[4];
  const float* b_Fc = (const float*)d_in[5];
  const float* W_Fh = (const float*)d_in[6];
  const float* b_Fh = (const float*)d_in[7];
  const float* W_Gc = (const float*)d_in[8];
  const float* b_Gc = (const float*)d_in[9];
  const float* W_Z  = (const float*)d_in[10];
  const float* b_Z  = (const float*)d_in[11];
  const float* W_A  = (const float*)d_in[12];
  const float* b_A  = (const float*)d_in[13];
  float* out = (float*)d_out;

  // Workspace layout (256B-aligned regions)
  char* ws = (char*)d_ws;
  __bf16* wcat = (__bf16*)(ws);                       // 5*131072 bf16 = 1310720 B
  float*  bcat = (float*)(ws + 1310720);              // 1024 f32     = 4096 B
  float*  deg  = (float*)(ws + 1314816);              // 5000 f32     (pad to 20224)
  float*  sum  = (float*)(ws + 1335040);              // 5000*2048 f32 = 40960000 B
  __bf16* ubf  = (__bf16*)(ws + 42295040);            // 40000*512 bf16 = 40960000 B

  // 0) weights/biases -> bf16 blob + packed bias
  k_convert_w<<<(5 * 131072 + 255) / 256, 256, 0, stream>>>(W_Fc, W_Fh, W_Gc, W_Z, W_A, wcat);
  k_convert_b<<<4, 256, 0, stream>>>(b_Fc, b_Fh, b_Gc, b_Z, bcat);

  // 1) zero deg + sum (contiguous region), count degrees, scatter-add h
  hipMemsetAsync(deg, 0, 20224 + (size_t)40960000, stream);
  k_degree<<<(EE + 255) / 256, 256, 0, stream>>>(dst, deg);
  k_scatter<<<EE, 256, 0, stream>>>(u, src, dst, sum);

  // 2) activation buffer: c part
  k_copy_c<<<(ROWS * PP + 255) / 256, 256, 0, stream>>>(u, ubf);

  // 3) stage-1 GEMM (W_A) -> h_ (bf16 into ubf)
  k_stage1<<<ROWS / 64, 256, 64 * 512 * sizeof(__bf16), stream>>>(
      u, sum, deg, wcat /* W_A at offset 0 */, b_A, ubf);

  // 4) stage-2 GEMM (Fc|Fh|Gc|Zt) + fused epilogue -> out
  k_stage2<<<ROWS / 16, 256, 16 * 1024 * sizeof(float), stream>>>(
      u, ubf, wcat + 131072 /* [W_Fc;W_Fh;W_Gc;W_Z] as 1024x512 */, bcat, out);
}
